// Translate_12558484373674
// MI455X (gfx1250) — matrix operand order repaired and checked by reading, not compiled, not run on hardware
//
#include <hip/hip_runtime.h>

// ---------------- types / constants ----------------
typedef __bf16 v16bf __attribute__((ext_vector_type(16)));
typedef float  v8f   __attribute__((ext_vector_type(8)));

#define VOC   32000
#define DIMD  512
#define HH    512
#define BB    32      // batch
#define KB    5       // beam
#define SRCL  32      // source length
#define TP1   34      // T+1 (T = SRCL+1 = 33)
#define ROWS  160     // BB*KB
#define KT    16      // 512 / 32 K-tiles
#define NEGF  (-1e30f)

// ---------------- helpers ----------------
__device__ __forceinline__ unsigned short f2bf(float f) {
  unsigned u = __float_as_uint(f);
  u += 0x7fffu + ((u >> 16) & 1u);           // round-to-nearest-even
  return (unsigned short)(u >> 16);
}
__device__ __forceinline__ float bf2f(unsigned short h) {
  return __uint_as_float(((unsigned)h) << 16);
}
__device__ __forceinline__ void split2(float v, unsigned short* hi, unsigned short* lo) {
  unsigned short h = f2bf(v);
  *hi = h;
  *lo = f2bf(v - bf2f(h));
}
__device__ __forceinline__ v16bf ld_tile(const unsigned short* base, int tile, int lane) {
  return *reinterpret_cast<const v16bf*>(base + (((size_t)tile) << 9) + (lane << 4));
}
__device__ __forceinline__ v8f wmma_bf(v16bf a, v16bf b, v8f c) {
  return __builtin_amdgcn_wmma_f32_16x16x32_bf16(false, a, false, b, (short)0, c, false, false);
}

// ---------------- init ----------------
__global__ void init_state(float* henc, int* seqA, float* best_ll, int* best_seq, int* last_tok) {
  int i = blockIdx.x * 256 + threadIdx.x;
  if (i < BB * HH)   henc[i] = 0.f;
  if (i < ROWS*TP1)  seqA[i] = 2;         // PAD
  if (i < BB)        best_ll[i] = NEGF;
  if (i < BB * TP1)  best_seq[i] = 2;     // PAD
  if (i < ROWS)      last_tok[i] = 0;     // SOS
}

// ---------------- weight pack (B operand, hi/lo bf16 split, WMMA lane layout) ----------
// B tile (kt,nt): n = lane&15 ; k = kt*32 + 8*(lane>>4) + (j<8 ? j : j+8).
// transposed=1: B[k][n] = W[n*ld + k]   (Wih/Whh, gh = h @ W^T)
// transposed=0: B[k][n] = W[k*ld + n]   (W_out)
__global__ void pack_w(const float* __restrict__ W, unsigned short* __restrict__ hi,
                       unsigned short* __restrict__ lo, int NT, int ld, int transposed) {
  int p = blockIdx.x * 256 + threadIdx.x;
  int j = p & 15, lane = (p >> 4) & 31, kt = (p >> 9) & 15, nt = p >> 13;
  if (nt >= NT) return;
  int n = nt * 16 + (lane & 15);
  int k = kt * 32 + ((lane >> 4) << 3) + (j < 8 ? j : j + 8);
  float v = transposed ? W[(size_t)n * ld + k] : W[(size_t)k * ld + n];
  unsigned short h, l; split2(v, &h, &l);
  hi[p] = h; lo[p] = l;
}

// ---------------- A-operand pack (hi/lo split) with row-gather modes -----------------
// A tile element: m = lane&15 ; k = kt*32 + 8*(lane>>4) + (j<8 ? j : j+8)
// mode 0: identity   mode 1: row = source[m*SRCL+t] (encoder x)
// mode 2: row = tok[m] (decoder x from emb_tgt)
// mode 3: row = (m/KB)*KB + parent[m] (hidden gather)
// mode 4: row = m/KB (broadcast h_enc to beams)
__global__ void pack_a(const float* __restrict__ src, const int* __restrict__ aux,
                       unsigned short* __restrict__ hi, unsigned short* __restrict__ lo,
                       float* __restrict__ fdst, int M, int mode, int t) {
  int p = blockIdx.x * 256 + threadIdx.x;
  int j = p & 15, lane = (p >> 4) & 31, kt = (p >> 9) & 15, mt = p >> 13;
  int m = mt * 16 + (lane & 15);
  int k = kt * 32 + ((lane >> 4) << 3) + (j < 8 ? j : j + 8);
  float v = 0.f;
  if (m < M) {
    int row;
    switch (mode) {
      case 0:  row = m; break;
      case 1:  row = aux[m * SRCL + t]; break;
      case 2:  row = aux[m]; break;
      case 3:  row = (m / KB) * KB + aux[m]; break;
      default: row = m / KB; break;
    }
    v = src[(size_t)row * DIMD + k];
    if (fdst) fdst[m * DIMD + k] = v;
  }
  unsigned short h, l; split2(v, &h, &l);
  hi[p] = h; lo[p] = l;
}

// ---------------- fused GRU cell: two GEMMs (x@Wih^T, h@Whh^T) + gates -------------
// grid.x = MT*32; one wave per block. Each block: 6 16x16 output tiles (r,z,n for gi,gh).
__global__ __launch_bounds__(32) void gru_gemm(
    const unsigned short* axh, const unsigned short* axl,
    const unsigned short* ahh, const unsigned short* ahl,
    const unsigned short* wxh, const unsigned short* wxl,
    const unsigned short* whh_, const unsigned short* whl,
    const float* __restrict__ bih, const float* __restrict__ bhh,
    const float* __restrict__ hold, float* __restrict__ hout) {
  int lane = threadIdx.x;
  int mt = blockIdx.x >> 5;
  int jt = blockIdx.x & 31;
  v8f zero = {0.f,0.f,0.f,0.f,0.f,0.f,0.f,0.f};
  v8f acc[6];
#pragma unroll
  for (int q = 0; q < 6; ++q) acc[q] = zero;
  for (int kt = 0; kt < KT; ++kt) {
    v16bf aXh = ld_tile(axh, mt * KT + kt, lane);
    v16bf aXl = ld_tile(axl, mt * KT + kt, lane);
    v16bf aHh = ld_tile(ahh, mt * KT + kt, lane);
    v16bf aHl = ld_tile(ahl, mt * KT + kt, lane);
#pragma unroll
    for (int g = 0; g < 3; ++g) {
      int bt = (g * 32 + jt) * KT + kt;
      v16bf bXh = ld_tile(wxh, bt, lane);
      v16bf bXl = ld_tile(wxl, bt, lane);
      v16bf bHh = ld_tile(whh_, bt, lane);
      v16bf bHl = ld_tile(whl, bt, lane);
      acc[g]     = wmma_bf(aXh, bXh, acc[g]);
      acc[g]     = wmma_bf(aXh, bXl, acc[g]);
      acc[g]     = wmma_bf(aXl, bXh, acc[g]);
      acc[3 + g] = wmma_bf(aHh, bHh, acc[3 + g]);
      acc[3 + g] = wmma_bf(aHh, bHl, acc[3 + g]);
      acc[3 + g] = wmma_bf(aHl, bHh, acc[3 + g]);
    }
  }
  int n  = jt * 16 + (lane & 15);
  int m0 = mt * 16 + ((lane >> 4) << 3);
  float br_i = bih[n], bz_i = bih[HH + n], bn_i = bih[2 * HH + n];
  float br_h = bhh[n], bz_h = bhh[HH + n], bn_h = bhh[2 * HH + n];
#pragma unroll
  for (int i = 0; i < 8; ++i) {
    int m = m0 + i;
    float r  = 1.f / (1.f + expf(-(acc[0][i] + br_i + acc[3][i] + br_h)));
    float z  = 1.f / (1.f + expf(-(acc[1][i] + bz_i + acc[4][i] + bz_h)));
    float nn = tanhf(acc[2][i] + bn_i + r * (acc[5][i] + bn_h));
    hout[m * HH + n] = (1.f - z) * nn + z * hold[m * HH + n];
  }
}

// ---------------- logits GEMM: h1(160x512) @ W_out(512x32000) + b_out -------------
// grid.x = 10 * 500 ; block handles 16x64 output tile (4 N-tiles), 3xBF16 split.
__global__ __launch_bounds__(32) void logits_gemm(
    const unsigned short* ah, const unsigned short* al,
    const unsigned short* wh, const unsigned short* wl,
    const float* __restrict__ bout, float* __restrict__ logits) {
  int lane = threadIdx.x;
  int mt  = blockIdx.x % 10;
  int nt0 = (blockIdx.x / 10) * 4;
  v8f zero = {0.f,0.f,0.f,0.f,0.f,0.f,0.f,0.f};
  v8f acc[4];
#pragma unroll
  for (int q = 0; q < 4; ++q) acc[q] = zero;
  for (int kt = 0; kt < KT; ++kt) {
    v16bf A  = ld_tile(ah, mt * KT + kt, lane);
    v16bf Al = ld_tile(al, mt * KT + kt, lane);
#pragma unroll
    for (int j = 0; j < 4; ++j) {
      int bt = (nt0 + j) * KT + kt;
      if (kt + 1 < KT) {                      // stream next K-tile of W_out
        __builtin_prefetch((const void*)(wh + (((size_t)(bt + 1)) << 9) + (lane << 4)), 0, 1);
      }
      v16bf Bh = ld_tile(wh, bt, lane);
      v16bf Bl = ld_tile(wl, bt, lane);
      acc[j] = wmma_bf(A,  Bh, acc[j]);
      acc[j] = wmma_bf(A,  Bl, acc[j]);
      acc[j] = wmma_bf(Al, Bh, acc[j]);
    }
  }
  int m0 = mt * 16 + ((lane >> 4) << 3);
#pragma unroll
  for (int j = 0; j < 4; ++j) {
    int n = (nt0 + j) * 16 + (lane & 15);
    float bo = bout[n];
#pragma unroll
    for (int i = 0; i < 8; ++i)
      logits[(size_t)(m0 + i) * VOC + n] = acc[j][i] + bo;
  }
}

// ---------------- per-row online log-sum-exp + top-5 (one pass) --------------------
__global__ __launch_bounds__(256) void row_reduce(const float* __restrict__ logits,
    float* __restrict__ lse, float* __restrict__ t5v, int* __restrict__ t5i) {
  __shared__ float sm[256], ss[256];
  __shared__ float cv[256 * 5];
  __shared__ int   ci[256 * 5];
  int row = blockIdx.x, tid = threadIdx.x;
  const float* rp = logits + (size_t)row * VOC;
  float m = -3.4e38f, s = 0.f;
  float tv[5] = {-3.4e38f, -3.4e38f, -3.4e38f, -3.4e38f, -3.4e38f};
  int   ti[5] = {-1, -1, -1, -1, -1};
  for (int c = tid; c < VOC; c += 256) {
    float v = rp[c];
    if (v > m) { s = s * expf(m - v) + 1.f; m = v; } else { s += expf(v - m); }
    if (v > tv[4]) {
      tv[4] = v; ti[4] = c;
      for (int q = 4; q > 0; --q)
        if (tv[q] > tv[q - 1]) {
          float fv = tv[q]; tv[q] = tv[q - 1]; tv[q - 1] = fv;
          int   fi = ti[q]; ti[q] = ti[q - 1]; ti[q - 1] = fi;
        }
    }
  }
  sm[tid] = m; ss[tid] = s;
  for (int q = 0; q < 5; ++q) { cv[tid * 5 + q] = tv[q]; ci[tid * 5 + q] = ti[q]; }
  __syncthreads();
  for (int st = 128; st > 0; st >>= 1) {
    if (tid < st) {
      float m2 = sm[tid + st], s2 = ss[tid + st];
      float M = fmaxf(sm[tid], m2);
      ss[tid] = ss[tid] * expf(sm[tid] - M) + s2 * expf(m2 - M);
      sm[tid] = M;
    }
    __syncthreads();
  }
  if (tid == 0) {
    lse[row] = sm[0] + logf(ss[0]);
    float bv[5] = {-3.4e38f, -3.4e38f, -3.4e38f, -3.4e38f, -3.4e38f};
    int   bi[5] = {-1, -1, -1, -1, -1};
    for (int e = 0; e < 256 * 5; ++e) {
      float v = cv[e]; int id = ci[e];
      if (id < 0) continue;
      if (v > bv[4] || (v == bv[4] && (unsigned)id < (unsigned)bi[4])) {
        bv[4] = v; bi[4] = id;
        for (int q = 4; q > 0; --q)
          if (bv[q] > bv[q - 1] || (bv[q] == bv[q - 1] && bi[q] < bi[q - 1])) {
            float fv = bv[q]; bv[q] = bv[q - 1]; bv[q - 1] = fv;
            int   fi = bi[q]; bi[q] = bi[q - 1]; bi[q - 1] = fi;
          }
      }
    }
    for (int q = 0; q < 5; ++q) { t5v[row * 5 + q] = bv[q]; t5i[row * 5 + q] = bi[q]; }
  }
}

// ---------------- step 0: initial_advance ----------------
__global__ void update0(const float* __restrict__ t5v, const int* __restrict__ t5i,
                        const float* __restrict__ lse, float* act_ll, int* last_tok,
                        int* parent, int* seqA) {
  int b = threadIdx.x; if (b >= BB) return;
  int row = b * KB;
  float L = lse[row];
  for (int k = 0; k < KB; ++k) {
    float v = t5v[row * 5 + k] - L;       // act_ll starts at 0
    int   w = t5i[row * 5 + k];
    act_ll[b * KB + k]   = v;
    last_tok[b * KB + k] = w;
    parent[b * KB + k]   = k;             // all beams identical after step 0
    seqA[(b * KB + k) * TP1 + 0] = 0;     // SOS
    seqA[(b * KB + k) * TP1 + 1] = w;
  }
}

// ---------------- beam update for step t (1..32) ----------------
__global__ void beam_update(const float* __restrict__ t5v, const int* __restrict__ t5i,
                            const float* __restrict__ lse, float* act_ll, int* last_tok,
                            int* parent, const int* __restrict__ seqIn, int* seqOut,
                            float* best_ll, int* best_seq, int t) {
  int b = threadIdx.x; if (b >= BB) return;
  float cv[25]; int cw[25];
  for (int k = 0; k < KB; ++k) {
    int row = b * KB + k;
    float base = act_ll[row] - lse[row];
    for (int i = 0; i < 5; ++i) {
      cv[k * 5 + i] = base + t5v[row * 5 + i];
      cw[k * 5 + i] = t5i[row * 5 + i];
    }
  }
  int sel[5]; bool used[25];
  for (int c = 0; c < 25; ++c) used[c] = false;
  for (int s = 0; s < 5; ++s) {
    float bv = -3.4e38f; int bi = 0;
    for (int c = 0; c < 25; ++c)
      if (!used[c] && cv[c] > bv) { bv = cv[c]; bi = c; }
    used[bi] = true; sel[s] = bi;
  }
  float cbest = NEGF; int carg = -1;
  for (int s = 0; s < 5; ++s) {
    int c = sel[s]; int p = c / 5; int w = cw[c]; float v = cv[c];
    parent[b * KB + s]   = p;
    last_tok[b * KB + s] = w;
    bool finite = v > -5e29f;
    bool eos = (w == 1) && finite;
    for (int j = 0; j < TP1; ++j) {
      int tok = (j == t + 1) ? w : seqIn[(b * KB + p) * TP1 + j];
      seqOut[(b * KB + s) * TP1 + j] = tok;
    }
    act_ll[b * KB + s] = (finite && !eos) ? v : NEGF;
    float comp = eos ? v / (float)(t + 2) : NEGF;
    if (comp > cbest) { cbest = comp; carg = s; }
  }
  if (cbest > best_ll[b]) {
    best_ll[b] = cbest;
    for (int j = 0; j < TP1; ++j)
      best_seq[b * TP1 + j] = seqOut[(b * KB + carg) * TP1 + j];
  }
}

// ---------------- winner ----------------
__global__ void finalize(const float* __restrict__ best_ll, const int* __restrict__ best_seq,
                         const float* __restrict__ act_ll, const int* __restrict__ seqFinal,
                         float* out) {
  int b = threadIdx.x; if (b >= BB) return;
  bool have = best_ll[b] > -5e29f;
  int bk = 0; float bv = act_ll[b * KB];
  for (int k = 1; k < KB; ++k)
    if (act_ll[b * KB + k] > bv) { bv = act_ll[b * KB + k]; bk = k; }
  for (int j = 0; j < TP1; ++j) {
    int tok = have ? best_seq[b * TP1 + j] : seqFinal[(b * KB + bk) * TP1 + j];
    out[b * TP1 + j] = (float)tok;
  }
}

// ---------------- host ----------------
extern "C" void kernel_launch(void* const* d_in, const int* in_sizes, int n_in,
                              void* d_out, int out_size, void* d_ws, size_t ws_size,
                              hipStream_t stream) {
  (void)in_sizes; (void)n_in; (void)out_size; (void)ws_size;
  const int*   source  = (const int*)d_in[0];
  const float* emb_src = (const float*)d_in[1];
  const float* emb_tgt = (const float*)d_in[2];
  const float* enc_Wih = (const float*)d_in[3];
  const float* enc_Whh = (const float*)d_in[4];
  const float* enc_bih = (const float*)d_in[5];
  const float* enc_bhh = (const float*)d_in[6];
  const float* dec_Wih = (const float*)d_in[7];
  const float* dec_Whh = (const float*)d_in[8];
  const float* dec_bih = (const float*)d_in[9];
  const float* dec_bhh = (const float*)d_in[10];
  const float* W_out   = (const float*)d_in[11];
  const float* b_out   = (const float*)d_in[12];
  float* out = (float*)d_out;

  char* w = (char*)d_ws;
  size_t off = 0;
  auto alloc = [&](size_t bytes) -> void* {
    void* p = (void*)(w + off);
    off += (bytes + 255) & ~((size_t)255);
    return p;
  };
  const size_t WOUT_PK = (size_t)2000 * KT * 512 * 2;   // bf16 elems * 2B
  const size_t GATE_PK = (size_t)96 * KT * 512 * 2;
  const size_t A_PK    = (size_t)10 * KT * 512 * 2;

  unsigned short* WoutH = (unsigned short*)alloc(WOUT_PK);
  unsigned short* WoutL = (unsigned short*)alloc(WOUT_PK);
  unsigned short* eWihH = (unsigned short*)alloc(GATE_PK);
  unsigned short* eWihL = (unsigned short*)alloc(GATE_PK);
  unsigned short* eWhhH = (unsigned short*)alloc(GATE_PK);
  unsigned short* eWhhL = (unsigned short*)alloc(GATE_PK);
  unsigned short* dWihH = (unsigned short*)alloc(GATE_PK);
  unsigned short* dWihL = (unsigned short*)alloc(GATE_PK);
  unsigned short* dWhhH = (unsigned short*)alloc(GATE_PK);
  unsigned short* dWhhL = (unsigned short*)alloc(GATE_PK);
  float* logits = (float*)alloc((size_t)ROWS * VOC * 4);
  float* hencA  = (float*)alloc((size_t)BB * HH * 4);
  float* hencB  = (float*)alloc((size_t)BB * HH * 4);
  float* hcur   = (float*)alloc((size_t)ROWS * HH * 4);
  float* h1     = (float*)alloc((size_t)ROWS * HH * 4);
  unsigned short* AxH = (unsigned short*)alloc(A_PK);
  unsigned short* AxL = (unsigned short*)alloc(A_PK);
  unsigned short* AhH = (unsigned short*)alloc(A_PK);
  unsigned short* AhL = (unsigned short*)alloc(A_PK);
  unsigned short* A1H = (unsigned short*)alloc(A_PK);
  unsigned short* A1L = (unsigned short*)alloc(A_PK);
  float* lse    = (float*)alloc(ROWS * 4);
  float* t5v    = (float*)alloc(ROWS * 5 * 4);
  int*   t5i    = (int*)alloc(ROWS * 5 * 4);
  float* act_ll = (float*)alloc(ROWS * 4);
  int*   last_tok = (int*)alloc(ROWS * 4);
  int*   parent   = (int*)alloc(ROWS * 4);
  int*   seqA   = (int*)alloc((size_t)ROWS * TP1 * 4);
  int*   seqB   = (int*)alloc((size_t)ROWS * TP1 * 4);
  float* best_ll  = (float*)alloc(BB * 4);
  int*   best_seq = (int*)alloc((size_t)BB * TP1 * 4);

  init_state<<<64, 256, 0, stream>>>(hencA, seqA, best_ll, best_seq, last_tok);

  // weight packs (hi/lo bf16 split, WMMA B-operand layout)
  pack_w<<<(2000 * KT * 512) / 256, 256, 0, stream>>>(W_out,   WoutH, WoutL, 2000, VOC, 0);
  pack_w<<<(96 * KT * 512) / 256,   256, 0, stream>>>(enc_Wih, eWihH, eWihL, 96, 512, 1);
  pack_w<<<(96 * KT * 512) / 256,   256, 0, stream>>>(enc_Whh, eWhhH, eWhhL, 96, 512, 1);
  pack_w<<<(96 * KT * 512) / 256,   256, 0, stream>>>(dec_Wih, dWihH, dWihL, 96, 512, 1);
  pack_w<<<(96 * KT * 512) / 256,   256, 0, stream>>>(dec_Whh, dWhhH, dWhhL, 96, 512, 1);

  // ---- encoder: 32 sequential GRU steps (B=32 -> MT=2) ----
  float* hin = hencA; float* hout = hencB;
  for (int t = 0; t < SRCL; ++t) {
    pack_a<<<64, 256, 0, stream>>>(emb_src, source, AxH, AxL, nullptr, BB, 1, t);
    pack_a<<<64, 256, 0, stream>>>(hin, nullptr, AhH, AhL, hcur, BB, 0, 0);
    gru_gemm<<<2 * 32, 32, 0, stream>>>(AxH, AxL, AhH, AhL,
                                        eWihH, eWihL, eWhhH, eWhhL,
                                        enc_bih, enc_bhh, hcur, hout);
    float* tmp = hin; hin = hout; hout = tmp;
  }
  // hin = final encoder hidden (32 x 512)

  // ---- decoder step 0 (initial_advance) ----
  pack_a<<<320, 256, 0, stream>>>(emb_tgt, last_tok, AxH, AxL, nullptr, ROWS, 2, 0);
  pack_a<<<320, 256, 0, stream>>>(hin, nullptr, AhH, AhL, hcur, ROWS, 4, 0);
  gru_gemm<<<10 * 32, 32, 0, stream>>>(AxH, AxL, AhH, AhL,
                                       dWihH, dWihL, dWhhH, dWhhL,
                                       dec_bih, dec_bhh, hcur, h1);
  pack_a<<<320, 256, 0, stream>>>(h1, nullptr, A1H, A1L, nullptr, ROWS, 0, 0);
  logits_gemm<<<10 * 500, 32, 0, stream>>>(A1H, A1L, WoutH, WoutL, b_out, logits);
  row_reduce<<<ROWS, 256, 0, stream>>>(logits, lse, t5v, t5i);
  update0<<<1, 32, 0, stream>>>(t5v, t5i, lse, act_ll, last_tok, parent, seqA);

  // ---- decoder steps t = 1 .. 32 ----
  int* sIn = seqA; int* sOut = seqB;
  for (int t = 1; t < TP1 - 1; ++t) {
    pack_a<<<320, 256, 0, stream>>>(emb_tgt, last_tok, AxH, AxL, nullptr, ROWS, 2, 0);
    pack_a<<<320, 256, 0, stream>>>(h1, parent, AhH, AhL, hcur, ROWS, 3, 0);
    gru_gemm<<<10 * 32, 32, 0, stream>>>(AxH, AxL, AhH, AhL,
                                         dWihH, dWihL, dWhhH, dWhhL,
                                         dec_bih, dec_bhh, hcur, h1);
    pack_a<<<320, 256, 0, stream>>>(h1, nullptr, A1H, A1L, nullptr, ROWS, 0, 0);
    logits_gemm<<<10 * 500, 32, 0, stream>>>(A1H, A1L, WoutH, WoutL, b_out, logits);
    row_reduce<<<ROWS, 256, 0, stream>>>(logits, lse, t5v, t5i);
    beam_update<<<1, 32, 0, stream>>>(t5v, t5i, lse, act_ll, last_tok, parent,
                                      sIn, sOut, best_ll, best_seq, t);
    int* tmp = sIn; sIn = sOut; sOut = tmp;
  }

  finalize<<<1, 32, 0, stream>>>(best_ll, best_seq, act_ll, sIn, out);
}
